// Attention_Key_Value_90984587199092
// MI455X (gfx1250) — matrix operand order repaired and checked by reading, not compiled, not run on hardware
//
#include <hip/hip_runtime.h>
#include <hip/hip_bf16.h>
#include <math.h>
#include <stdint.h>

typedef __attribute__((ext_vector_type(16))) _Float16 v16h;
typedef __attribute__((ext_vector_type(8)))  float    v8f;
typedef unsigned int u32x4 __attribute__((ext_vector_type(4)));
typedef int          i32x4 __attribute__((ext_vector_type(4)));
typedef int          i32x8 __attribute__((ext_vector_type(8)));

union Frag16 {
    v16h v;
    unsigned int u[8];
};

// problem constants
constexpr int cB = 2, cDIM = 512, cH = 128, cW = 128, cHEADS = 8, cC = 64;
constexpr int cHW = cH * cW; // 16384

// ---------------------------------------------------------------------------
// CDNA5 async data-movement helpers (guarded; fall back to plain copies)
// ---------------------------------------------------------------------------
#if __has_builtin(__builtin_amdgcn_global_load_async_to_lds_b64)
#define HAVE_ASYNC_LDS 1
// Per clang diagnostic: param0 is int2* in AS1 (__device__), param1 int2* in AS3.
typedef int a_v2i __attribute__((vector_size(8)));
typedef __attribute__((address_space(1))) a_v2i* async_gptr;
typedef __attribute__((address_space(3))) a_v2i* async_lptr;
__device__ __forceinline__ void async_ld_b64(void* lds, const void* g) {
    __builtin_amdgcn_global_load_async_to_lds_b64(
        (async_gptr)(uintptr_t)g, (async_lptr)(uintptr_t)lds, 0, 0);
}
__device__ __forceinline__ void wait_async0() {
#if __has_builtin(__builtin_amdgcn_s_wait_asynccnt)
    __builtin_amdgcn_s_wait_asynccnt(0);
#else
    asm volatile("s_wait_asynccnt 0x0" ::: "memory");
#endif
}
#endif

#if __has_builtin(__builtin_amdgcn_tensor_load_to_lds)
#define HAVE_TDM 1
// 2D TDM load: contiguous [rows x cols] f16 tile (stride == cols) -> LDS.
__device__ __forceinline__ void tdm_load_tile_f16(void* lds, const void* gsrc,
                                                  int rows, int cols) {
    uintptr_t ga = (uintptr_t)gsrc;
    unsigned  la = (unsigned)(uintptr_t)lds; // AS3 offset lives in low 32 bits
    u32x4 g0;
    g0[0] = 1u;                                            // count=1, user mode
    g0[1] = la;                                            // lds_addr
    g0[2] = (unsigned)(ga & 0xFFFFFFFFu);                  // global_addr[31:0]
    g0[3] = (unsigned)((ga >> 32) & 0x1FFFFFFu) | (2u << 30); // addr hi | type=2
    i32x8 g1;
    g1[0] = 1 << 16;                                       // data_size=2 bytes
    g1[1] = (cols & 0xFFFF) << 16;                         // tensor_dim0[15:0]
    g1[2] = ((cols >> 16) & 0xFFFF) | ((rows & 0xFFFF) << 16); // d0 hi | tensor_dim1
    g1[3] = (cols & 0xFFFF) << 16;                         // tile_dim0
    g1[4] = rows & 0xFFFF;                                 // tile_dim1 (tile_dim2=0)
    g1[5] = cols;                                          // tensor_dim0_stride lo
    g1[6] = 0;
    g1[7] = 0;
    i32x4 z4 = {0, 0, 0, 0};
#if __clang_major__ >= 23
    i32x8 z8 = {0, 0, 0, 0, 0, 0, 0, 0};
    __builtin_amdgcn_tensor_load_to_lds(g0, g1, z4, z4, z8, 0);
#else
    __builtin_amdgcn_tensor_load_to_lds(g0, g1, z4, z4, 0);
#endif
}
__device__ __forceinline__ void wait_tensor0() {
    __builtin_amdgcn_s_wait_tensorcnt(0);
}
#endif

// ---------------------------------------------------------------------------
// f32 -> f16 convert
// ---------------------------------------------------------------------------
__global__ __launch_bounds__(256) void cvt_f32_f16(const float* __restrict__ in,
                                                   _Float16* __restrict__ out,
                                                   size_t n) {
    size_t i = (size_t)blockIdx.x * 256 + threadIdx.x;
    if (i < n) out[i] = (_Float16)in[i];
}

// ---------------------------------------------------------------------------
// Generic pointwise-conv GEMM:
//   Out[b, o, p] = sum_c W[o, c] * In[b, c, p]
// W: f32 [O, Cin]; In: f16 [Batch, Cin, P]; Out: f16 or f32 [Batch, O, P]
// Block: 128 threads (4 waves). Tile: 64(o) x 64(p), K-step 32.
// ---------------------------------------------------------------------------
template <bool OUT_F32>
__global__ __launch_bounds__(128) void wmma_pw_gemm(
    const float* __restrict__ Wm, const _Float16* __restrict__ In,
    void* __restrict__ Out, int O, int Cin, int P) {
    __shared__ _Float16 Alds[64][34]; // [o_local][k_local]
    __shared__ _Float16 Blds[64][34]; // [p_local][k_local] (transposed stage)

    const int t    = threadIdx.x;
    const int lane = t & 31;
    const int wave = t >> 5;
    const int lm   = lane & 15;
    const int half = lane >> 4;
    const int p0   = blockIdx.x * 64;
    const int o0   = blockIdx.y * 64;
    const int b    = blockIdx.z;
    const size_t inBase = (size_t)b * Cin * (size_t)P;

    v8f acc[4];
    for (int i = 0; i < 4; ++i)
        for (int j = 0; j < 8; ++j) acc[i][j] = 0.0f;

    for (int k0 = 0; k0 < Cin; k0 += 32) {
        // Stage A (weights, f32->f16): 64x32
        for (int i = 0; i < 16; ++i) {
            int idx = t + 128 * i;            // 0..2047
            int r = idx >> 5, c = idx & 31;
            Alds[r][c] = (_Float16)Wm[(size_t)(o0 + r) * Cin + (k0 + c)];
        }
        // Stage B (activations, f16), transposed into [p][c]: 32x64
        for (int i = 0; i < 16; ++i) {
            int idx = t + 128 * i;            // 0..2047
            int c = idx >> 6, p = idx & 63;   // p fastest -> coalesced global read
            Blds[p][c] = In[inBase + (size_t)(k0 + c) * P + (p0 + p)];
        }
        if (k0 + 32 < Cin) {
            __builtin_prefetch(&Wm[(size_t)(o0 + (t >> 1)) * Cin + k0 + 32], 0, 0);
            __builtin_prefetch(&In[inBase + (size_t)(k0 + 32 + (t >> 2)) * P + p0], 0, 0);
        }
        __syncthreads();

        Frag16 a;
        const int ml = wave * 16 + lm;
        for (int j = 0; j < 8; ++j) {
            int kb = half * 16 + j * 2;
            a.u[j] = *(const unsigned int*)&Alds[ml][kb];
        }
        for (int pt = 0; pt < 4; ++pt) {
            Frag16 bf;
            const int nl = pt * 16 + lm;
            for (int j = 0; j < 8; ++j) {
                int kb = half * 16 + j * 2;
                bf.u[j] = *(const unsigned int*)&Blds[nl][kb];
            }
            acc[pt] = __builtin_amdgcn_wmma_f32_16x16x32_f16(
                false, a.v, false, bf.v, (short)0, acc[pt], false, false);
        }
        __syncthreads();
    }

    for (int pt = 0; pt < 4; ++pt) {
        int p = p0 + pt * 16 + lm;
        for (int j = 0; j < 8; ++j) {
            int o = o0 + wave * 16 + half * 8 + j;
            size_t oi = ((size_t)b * O + o) * (size_t)P + p;
            if (OUT_F32) ((float*)Out)[oi] = acc[pt][j];
            else         ((_Float16*)Out)[oi] = (_Float16)acc[pt][j];
        }
    }
}

// ---------------------------------------------------------------------------
// Depthwise 3x3, SAME padding: in/out f16 [B, CH, H, W], w f32 [CH, 9]
// ---------------------------------------------------------------------------
__global__ __launch_bounds__(256) void dwconv3x3_f16(
    const _Float16* __restrict__ in, const float* __restrict__ w,
    _Float16* __restrict__ out, int CH, size_t total) {
    size_t gid = (size_t)blockIdx.x * 256 + threadIdx.x;
    if (gid >= total) return;
    int pix = (int)(gid & (cHW - 1));
    size_t bc = gid >> 14;             // b*CH + ch
    int ch = (int)(bc % CH);
    int x = pix & (cW - 1), y = pix >> 7;
    const _Float16* base = in + bc * (size_t)cHW;
    float acc = 0.0f;
    #pragma unroll
    for (int dy = -1; dy <= 1; ++dy) {
        int yy = y + dy;
        if ((unsigned)yy >= (unsigned)cH) continue;
        #pragma unroll
        for (int dx = -1; dx <= 1; ++dx) {
            int xx = x + dx;
            if ((unsigned)xx >= (unsigned)cW) continue;
            acc += w[ch * 9 + (dy + 1) * 3 + (dx + 1)] * (float)base[yy * cW + xx];
        }
    }
    out[gid] = (_Float16)acc;
}

// ---------------------------------------------------------------------------
// cat([qkv-slice f16, feature f32]) -> depthwise 3x3 -> exact GELU -> f16
// ---------------------------------------------------------------------------
__global__ __launch_bounds__(256) void dwconv_cat_gelu(
    const _Float16* __restrict__ qkv, int chOff,
    const float* __restrict__ feat, const float* __restrict__ w,
    _Float16* __restrict__ out, size_t total) {
    size_t gid = (size_t)blockIdx.x * 256 + threadIdx.x;
    if (gid >= total) return;
    int pix = (int)(gid & (cHW - 1));
    size_t bc = gid >> 14;                 // b*(2*DIM) + ch
    int ch = (int)(bc % (2 * cDIM));
    int b  = (int)(bc / (2 * cDIM));
    int x = pix & (cW - 1), y = pix >> 7;
    float acc = 0.0f;
    #pragma unroll
    for (int dy = -1; dy <= 1; ++dy) {
        int yy = y + dy;
        if ((unsigned)yy >= (unsigned)cH) continue;
        #pragma unroll
        for (int dx = -1; dx <= 1; ++dx) {
            int xx = x + dx;
            if ((unsigned)xx >= (unsigned)cW) continue;
            float val;
            if (ch < cDIM)
                val = (float)qkv[((size_t)b * (3 * cDIM) + chOff + ch) * cHW + yy * cW + xx];
            else
                val = feat[((size_t)b * cDIM + (ch - cDIM)) * (size_t)cHW + yy * cW + xx];
            acc += w[ch * 9 + (dy + 1) * 3 + (dx + 1)] * val;
        }
    }
    float g = 0.5f * acc * (1.0f + erff(acc * 0.70710678118654752f));
    out[gid] = (_Float16)g;
}

// ---------------------------------------------------------------------------
// Per-channel L2 normalize over HW
// ---------------------------------------------------------------------------
__global__ __launch_bounds__(256) void l2norm_rows(_Float16* buf, int chanStride, int nchan) {
    __shared__ float red[256];
    int rid = blockIdx.x;
    int b = rid / nchan, c = rid % nchan;
    _Float16* row = buf + ((size_t)b * chanStride + c) * (size_t)cHW;
    int t = threadIdx.x;
    float s = 0.0f;
    for (int i = t; i < cHW; i += 256) {
        float v = (float)row[i];
        s += v * v;
    }
    red[t] = s;
    __syncthreads();
    for (int stride = 128; stride > 0; stride >>= 1) {
        if (t < stride) red[t] += red[t + stride];
        __syncthreads();
    }
    float inv = 1.0f / fmaxf(sqrtf(red[0]), 1e-12f);
    for (int i = t; i < cHW; i += 256) row[i] = (_Float16)((float)row[i] * inv);
}

// ---------------------------------------------------------------------------
// Score GEMM: S[b,h] (64x64) = Qn(64x16384) * Kn(64x16384)^T, * temperature[h]
// Q/K tiles are row-major contiguous -> staged with async global->LDS copies.
// LDS rows padded to 36 halfwords so 8B async chunks stay 8B-aligned.
// ---------------------------------------------------------------------------
__global__ __launch_bounds__(128) void wmma_score(
    const _Float16* __restrict__ qn, const _Float16* __restrict__ kn,
    const float* __restrict__ temperature, float* __restrict__ S) {
    __shared__ _Float16 Alds[64][36];
    __shared__ _Float16 Blds[64][36];
    const int bh = blockIdx.x;
    const int b = bh >> 3, h = bh & 7;
    const int t = threadIdx.x, lane = t & 31, wave = t >> 5;
    const int lm = lane & 15, half = lane >> 4;
    const size_t qBase = ((size_t)b * (3 * cDIM) + h * cC) * (size_t)cHW;
    const size_t kBase = ((size_t)b * cDIM + h * cC) * (size_t)cHW;

    v8f acc[4];
    for (int i = 0; i < 4; ++i)
        for (int j = 0; j < 8; ++j) acc[i][j] = 0.0f;

    for (int k0 = 0; k0 < cHW; k0 += 32) {
#ifdef HAVE_ASYNC_LDS
        for (int i = 0; i < 4; ++i) {
            int idx = t + 128 * i;     // 0..511 -> 64 rows x 8 chunks of 8B
            int r = idx >> 3, sub = idx & 7;
            async_ld_b64(&Alds[r][sub * 4], qn + qBase + (size_t)r * cHW + k0 + sub * 4);
            async_ld_b64(&Blds[r][sub * 4], kn + kBase + (size_t)r * cHW + k0 + sub * 4);
        }
        wait_async0();
#else
        for (int i = 0; i < 16; ++i) {
            int idx = t + 128 * i;
            int r = idx >> 5, c = idx & 31;
            Alds[r][c] = qn[qBase + (size_t)r * cHW + k0 + c];
            Blds[r][c] = kn[kBase + (size_t)r * cHW + k0 + c];
        }
#endif
        __syncthreads();
        Frag16 a;
        const int ml = wave * 16 + lm;
        for (int j = 0; j < 8; ++j) {
            int kb = half * 16 + j * 2;
            a.u[j] = *(const unsigned int*)&Alds[ml][kb];
        }
        for (int pt = 0; pt < 4; ++pt) {
            Frag16 bf;
            const int nl = pt * 16 + lm;
            for (int j = 0; j < 8; ++j) {
                int kb = half * 16 + j * 2;
                bf.u[j] = *(const unsigned int*)&Blds[nl][kb];
            }
            acc[pt] = __builtin_amdgcn_wmma_f32_16x16x32_f16(
                false, a.v, false, bf.v, (short)0, acc[pt], false, false);
        }
        __syncthreads();
    }
    float tv = temperature[h];
    for (int pt = 0; pt < 4; ++pt) {
        int n = pt * 16 + lm;
        for (int j = 0; j < 8; ++j) {
            int m = wave * 16 + half * 8 + j;
            S[((size_t)bh * cC + m) * cC + n] = acc[pt][j] * tv;
        }
    }
}

// ---------------------------------------------------------------------------
// Top-k mask + softmax per row of 64; kcount from device scalar K.
// ---------------------------------------------------------------------------
__global__ __launch_bounds__(64) void topk_softmax(
    const float* __restrict__ S, const int* __restrict__ Kp,
    _Float16* __restrict__ Ah) {
    __shared__ float vals[64];
    __shared__ float red[64];
    const int row = blockIdx.x;
    const int i = threadIdx.x;
    float v = S[(size_t)row * 64 + i];
    vals[i] = v;
    __syncthreads();
    const int kcount = (cC * (*Kp)) / 100;
    int rank = 0;
    for (int j = 0; j < 64; ++j) {
        float o = vals[j];
        rank += (o > v) || (o == v && j < i);
    }
    const bool keep = rank < kcount;
    red[i] = keep ? v : -3.4e38f;
    __syncthreads();
    for (int s = 32; s > 0; s >>= 1) {
        if (i < s) red[i] = fmaxf(red[i], red[i + s]);
        __syncthreads();
    }
    float mx = red[0];
    __syncthreads();
    float e = keep ? __expf(v - mx) : 0.0f;
    red[i] = e;
    __syncthreads();
    for (int s = 32; s > 0; s >>= 1) {
        if (i < s) red[i] += red[i + s];
        __syncthreads();
    }
    Ah[(size_t)row * 64 + i] = (_Float16)(e / red[0]);
}

// ---------------------------------------------------------------------------
// Attention-V GEMM: Out[b,h] (64 x 16384) = Attn(64x64) * V(64x16384), K=64.
// The 8KB attn tile is fetched with one TDM tensor_load_to_lds (wave 0).
// ---------------------------------------------------------------------------
__global__ __launch_bounds__(128) void wmma_av(
    const _Float16* __restrict__ Ah, const _Float16* __restrict__ V,
    _Float16* __restrict__ Out) {
    __shared__ _Float16 Alds[64][64]; // [m][k] full 64x64 attn (unpadded: TDM row-major)
    __shared__ _Float16 Blds[64][66]; // [n][k] transposed V tile
    const int n0 = blockIdx.x * 64;
    const int bh = blockIdx.y;
    const int b = bh >> 3, h = bh & 7;
    const int t = threadIdx.x, lane = t & 31, wave = t >> 5;
    const int lm = lane & 15, half = lane >> 4;
    const size_t vBase = ((size_t)b * cDIM + h * cC) * (size_t)cHW;

#ifdef HAVE_TDM
    if (wave == 0) {
        tdm_load_tile_f16(&Alds[0][0], Ah + (size_t)bh * 4096, 64, 64);
        wait_tensor0();
    }
#else
    for (int i = 0; i < 32; ++i) {
        int idx = t + 128 * i;
        int r = idx >> 6, c = idx & 63;
        Alds[r][c] = Ah[(size_t)bh * 4096 + r * 64 + c];
    }
#endif
    for (int i = 0; i < 32; ++i) {
        int idx = t + 128 * i;           // 0..4095
        int r = idx >> 6, c = idx & 63;  // k = r, n = c -> coalesced global read
        Blds[c][r] = V[vBase + (size_t)r * cHW + n0 + c];
    }
    __syncthreads();

    v8f acc[4];
    for (int i = 0; i < 4; ++i)
        for (int j = 0; j < 8; ++j) acc[i][j] = 0.0f;

    const int ml = wave * 16 + lm;
    for (int ks = 0; ks < 64; ks += 32) {
        Frag16 a;
        for (int j = 0; j < 8; ++j) {
            int kb = ks + half * 16 + j * 2;
            a.u[j] = *(const unsigned int*)&Alds[ml][kb];
        }
        for (int pt = 0; pt < 4; ++pt) {
            Frag16 bf;
            const int nl = pt * 16 + lm;
            for (int j = 0; j < 8; ++j) {
                int kb = ks + half * 16 + j * 2;
                bf.u[j] = *(const unsigned int*)&Blds[nl][kb];
            }
            acc[pt] = __builtin_amdgcn_wmma_f32_16x16x32_f16(
                false, a.v, false, bf.v, (short)0, acc[pt], false, false);
        }
    }
    for (int pt = 0; pt < 4; ++pt) {
        int n = n0 + pt * 16 + lm;
        for (int j = 0; j < 8; ++j) {
            int m = wave * 16 + half * 8 + j;
            Out[((size_t)b * cDIM + h * cC + m) * (size_t)cHW + n] = (_Float16)acc[pt][j];
        }
    }
}

// ---------------------------------------------------------------------------
// Host-side launcher
// ---------------------------------------------------------------------------
extern "C" void kernel_launch(void* const* d_in, const int* in_sizes, int n_in,
                              void* d_out, int out_size, void* d_ws, size_t ws_size,
                              hipStream_t stream) {
    (void)in_sizes; (void)n_in; (void)out_size; (void)ws_size;
    const float* x        = (const float*)d_in[0];
    const float* f1       = (const float*)d_in[1];
    const float* f2       = (const float*)d_in[2];
    const float* temp     = (const float*)d_in[3];
    const float* w_qkv    = (const float*)d_in[4];
    const float* w_qkv_dw = (const float*)d_in[5];
    const float* w_k_dw   = (const float*)d_in[6];
    const float* w_k_pw   = (const float*)d_in[7];
    const float* w_v_dw   = (const float*)d_in[8];
    const float* w_v_pw   = (const float*)d_in[9];
    const float* w_proj   = (const float*)d_in[10];
    const int*   Kp       = (const int*)d_in[11];

    char* ws = (char*)d_ws;
    _Float16* xh   = (_Float16*)(ws + 0);            // 32 MB  x as f16
    _Float16* t0   = (_Float16*)(ws + 33554432ull);  // 96 MB  pwconv(qkv) pre-dw
    _Float16* cat  = (_Float16*)(ws + 33554432ull);  // 64 MB  reuse of t0 region
    _Float16* qkv  = (_Float16*)(ws + 134217728ull); // 96 MB  dwconv output
    _Float16* kbuf = (_Float16*)(ws + 234881024ull); // 32 MB
    _Float16* vbuf = (_Float16*)(ws + 268435456ull); // 32 MB
    float*    S    = (float*)   (ws + 301989888ull); // 256 KB scores
    _Float16* ah   = (_Float16*)(ws + 302252032ull); // 128 KB attn f16
    _Float16* oav  = (_Float16*)(ws + 302383104ull); // 32 MB  attn@V

    const size_t nX = (size_t)cB * cDIM * cHW;
    cvt_f32_f16<<<dim3((unsigned)((nX + 255) / 256)), 256, 0, stream>>>(x, xh, nX);

    wmma_pw_gemm<false><<<dim3(cHW / 64, (3 * cDIM) / 64, cB), 128, 0, stream>>>(
        w_qkv, xh, (void*)t0, 3 * cDIM, cDIM, cHW);
    {
        size_t n = (size_t)cB * 3 * cDIM * cHW;
        dwconv3x3_f16<<<dim3((unsigned)((n + 255) / 256)), 256, 0, stream>>>(
            t0, w_qkv_dw, qkv, 3 * cDIM, n);
    }
    {
        size_t n = (size_t)cB * 2 * cDIM * cHW;
        dwconv_cat_gelu<<<dim3((unsigned)((n + 255) / 256)), 256, 0, stream>>>(
            qkv, cDIM, f1, w_k_dw, cat, n);
        wmma_pw_gemm<false><<<dim3(cHW / 64, cDIM / 64, cB), 128, 0, stream>>>(
            w_k_pw, cat, (void*)kbuf, cDIM, 2 * cDIM, cHW);
        dwconv_cat_gelu<<<dim3((unsigned)((n + 255) / 256)), 256, 0, stream>>>(
            qkv, 2 * cDIM, f2, w_v_dw, cat, n);
        wmma_pw_gemm<false><<<dim3(cHW / 64, cDIM / 64, cB), 128, 0, stream>>>(
            w_v_pw, cat, (void*)vbuf, cDIM, 2 * cDIM, cHW);
    }

    l2norm_rows<<<dim3(cB * cDIM), 256, 0, stream>>>(qkv, 3 * cDIM, cDIM);
    l2norm_rows<<<dim3(cB * cDIM), 256, 0, stream>>>(kbuf, cDIM, cDIM);

    wmma_score<<<dim3(cB * cHEADS), 128, 0, stream>>>(qkv, kbuf, temp, S);
    topk_softmax<<<dim3(cB * cHEADS * cC), 64, 0, stream>>>(S, Kp, ah);

    wmma_av<<<dim3(cHW / 64, cB * cHEADS), 128, 0, stream>>>(ah, vbuf, oav);
    wmma_pw_gemm<true><<<dim3(cHW / 64, cDIM / 64, cB), 128, 0, stream>>>(
        w_proj, oav, d_out, cDIM, cDIM, cHW);
}
